// ScanLSTM_42580305773031
// MI455X (gfx1250) — matrix operand order repaired and checked by reading, not compiled, not run on hardware
//
#include <hip/hip_runtime.h>

// ---------------------------------------------------------------------------
// ScanLSTM for MI455X (gfx1250, wave32, WMMA).
// Gate order [i, f, g, o].  Z = x@Wi + b precomputed (parallel GEMM),
// recurrent h@Wh done per-step with v_wmma_f32_16x16x32_bf16, fp32 accumulate.
// ---------------------------------------------------------------------------

typedef __attribute__((ext_vector_type(8)))  float  v8f;
typedef __attribute__((ext_vector_type(16))) __bf16 v16bf;
typedef __attribute__((ext_vector_type(8)))  __bf16 v8bf;

#define T_STEPS 128
#define B_DIM   64
#define D_DIM   1024
#define H_DIM   1024
#define G_DIM   4096   // 4*H

// f32 -> bf16 via native convert (v_cvt_pk_bf16_f32 class, RNE on gfx1250)
__device__ __forceinline__ __bf16 f2bf(float f) { return (__bf16)f; }

// packed 8-wide f32 -> bf16 conversion; backend selects packed cvt ops
__device__ __forceinline__ v8bf cvt8(v8f f) {
    return __builtin_convertvector(f, v8bf);
}
__device__ __forceinline__ v16bf join16(v8bf lo, v8bf hi) {
    return __builtin_shufflevector(lo, hi, 0, 1, 2, 3, 4, 5, 6, 7,
                                   8, 9, 10, 11, 12, 13, 14, 15);
}

__device__ __forceinline__ float sigf(float x) {
    return 1.0f / (1.0f + __expf(-x));
}
// NaN-safe tanh via exp of a non-positive argument (no inf/inf)
__device__ __forceinline__ float tanhf_fast(float x) {
    float ax = fabsf(x);
    float e  = __expf(-2.0f * ax);
    float t  = (1.0f - e) / (1.0f + e);
    return copysignf(t, x);
}

// ---------------------------------------------------------------------------
// W [K=1024][N=4096] fp32 (row-major)  ->  Wt [N=4096][K=1024] bf16
// LDS-tiled so both global read and write are coalesced.
// grid (N/16, K/16), block (16,16)
// ---------------------------------------------------------------------------
__global__ void wt_transpose(const float* __restrict__ W, __bf16* __restrict__ Wt) {
    __shared__ __bf16 tile[16][17];
    const int n0 = blockIdx.x << 4;
    const int k0 = blockIdx.y << 4;
    const int tx = threadIdx.x, ty = threadIdx.y;
    tile[ty][tx] = f2bf(W[(size_t)(k0 + ty) * G_DIM + (n0 + tx)]);
    __syncthreads();
    Wt[(size_t)(n0 + ty) * D_DIM + (k0 + tx)] = tile[tx][ty];
}

// copy h0/c0 into state buffers (65536 threads)
__global__ void init_state(const float* __restrict__ h0, const float* __restrict__ c0,
                           float* __restrict__ h, float* __restrict__ c) {
    const int i = blockIdx.x * blockDim.x + threadIdx.x;
    h[i] = h0[i];
    c[i] = c0[i];
}

// ---------------------------------------------------------------------------
// Z = X @ Wit^T + b      X: [8192,1024] fp32 (converted in-register to bf16)
//                        Wit: [4096][1024] bf16 (column-major weight)
//                        Z:  [8192,4096] fp32
// One wave -> 16x64 output strip (4 accumulators, A fragment reused 4x).
// block = 128 threads (4 waves), grid = 8192 blocks -> 32768 waves.
// ---------------------------------------------------------------------------
__global__ void gemm_xwi(const float* __restrict__ X,
                         const __bf16* __restrict__ Wit,
                         const float* __restrict__ bias,
                         float* __restrict__ Z) {
    const int lane   = threadIdx.x & 31;
    const int wid    = blockIdx.x * 4 + (threadIdx.x >> 5);
    const int mtile  = wid >> 6;   // 0..511
    const int nstrip = wid & 63;   // 0..63
    const int l16    = lane & 15;
    const int hi     = (lane >> 4) & 1;

    // A (16x32 bf16) lane layout: lanes 0-15 -> K 0..15, lanes 16-31 -> K 16..31
    const float* arow = X + (size_t)(mtile * 16 + l16) * D_DIM + (hi ? 16 : 0);
    const int bkoff = hi ? 16 : 0;  // B lanes 16-31 hold K+16

    int           ncol[4];
    const __bf16* bptr[4];
#pragma unroll
    for (int j = 0; j < 4; ++j) {
        ncol[j] = nstrip * 64 + j * 16 + l16;
        bptr[j] = Wit + (size_t)ncol[j] * D_DIM + bkoff;
    }

    v8f acc0 = {}, acc1 = {}, acc2 = {}, acc3 = {};

    for (int kk = 0; kk < D_DIM; kk += 32) {
        v8f flo = *reinterpret_cast<const v8f*>(arow + kk);
        v8f fhi = *reinterpret_cast<const v8f*>(arow + kk + 8);
        v16bf a = join16(cvt8(flo), cvt8(fhi));

        v16bf b0 = *reinterpret_cast<const v16bf*>(bptr[0] + kk);
        v16bf b1 = *reinterpret_cast<const v16bf*>(bptr[1] + kk);
        v16bf b2 = *reinterpret_cast<const v16bf*>(bptr[2] + kk);
        v16bf b3 = *reinterpret_cast<const v16bf*>(bptr[3] + kk);

        acc0 = __builtin_amdgcn_wmma_f32_16x16x32_bf16(false, a, false, b0, (short)0, acc0, false, false);
        acc1 = __builtin_amdgcn_wmma_f32_16x16x32_bf16(false, a, false, b1, (short)0, acc1, false, false);
        acc2 = __builtin_amdgcn_wmma_f32_16x16x32_bf16(false, a, false, b2, (short)0, acc2, false, false);
        acc3 = __builtin_amdgcn_wmma_f32_16x16x32_bf16(false, a, false, b3, (short)0, acc3, false, false);
    }

    // C/D layout: VGPR v, lane l -> M = v + 8*(l>=16), N = l&15
#pragma unroll
    for (int v = 0; v < 8; ++v) {
        const size_t mrow = (size_t)(mtile * 16 + v + (hi ? 8 : 0));
        float* zr = Z + mrow * G_DIM;
        zr[ncol[0]] = acc0[v] + bias[ncol[0]];
        zr[ncol[1]] = acc1[v] + bias[ncol[1]];
        zr[ncol[2]] = acc2[v] + bias[ncol[2]];
        zr[ncol[3]] = acc3[v] + bias[ncol[3]];
    }
}

// ---------------------------------------------------------------------------
// One LSTM step:  z = Z[t] + mask(h_prev) @ Wh ;  gate fusion; update c, h.
// One wave -> one 16x16 (batch x hidden) tile, computing all 4 gates so the
// nonlinearity fuses on the WMMA accumulator layout.
// block = 128 threads (4 waves), grid = 64 blocks -> 256 waves (4 mtiles x 64 ntiles)
// ---------------------------------------------------------------------------
__global__ void lstm_step(const float* __restrict__ Z_t,          // [64,4096]
                          const unsigned char* __restrict__ rst,  // [64] bools
                          const float* __restrict__ h_prev,       // [64,1024]
                          const __bf16* __restrict__ Wht,         // [4096][1024]
                          float* __restrict__ c_state,            // [64,1024]
                          float* __restrict__ h_next,             // [64,1024]
                          float* __restrict__ ys_t) {             // [64,1024]
    const int lane  = threadIdx.x & 31;
    const int wid   = blockIdx.x * 4 + (threadIdx.x >> 5);
    const int mtile = wid >> 6;   // 0..3
    const int ntile = wid & 63;   // 0..63
    const int l16   = lane & 15;
    const int hi    = (lane >> 4) & 1;

    const int   arow  = mtile * 16 + l16;              // batch row this lane feeds into A
    const float amask = rst[arow] ? 0.0f : 1.0f;       // reset zeroes carried h
    const float* ap   = h_prev + (size_t)arow * H_DIM + (hi ? 16 : 0);
    const int   bkoff = hi ? 16 : 0;
    const int   col   = ntile * 16 + l16;              // hidden column

    const __bf16* bptr[4];
#pragma unroll
    for (int g = 0; g < 4; ++g)  // gate g's weight column = g*H + col
        bptr[g] = Wht + (size_t)(g * H_DIM + col) * D_DIM + bkoff;

    v8f acc0 = {}, acc1 = {}, acc2 = {}, acc3 = {};

    for (int kk = 0; kk < H_DIM; kk += 32) {
        v8f flo = *reinterpret_cast<const v8f*>(ap + kk);
        v8f fhi = *reinterpret_cast<const v8f*>(ap + kk + 8);
        v16bf a = join16(cvt8(flo * amask), cvt8(fhi * amask));

        v16bf b0 = *reinterpret_cast<const v16bf*>(bptr[0] + kk);
        v16bf b1 = *reinterpret_cast<const v16bf*>(bptr[1] + kk);
        v16bf b2 = *reinterpret_cast<const v16bf*>(bptr[2] + kk);
        v16bf b3 = *reinterpret_cast<const v16bf*>(bptr[3] + kk);

        acc0 = __builtin_amdgcn_wmma_f32_16x16x32_bf16(false, a, false, b0, (short)0, acc0, false, false);
        acc1 = __builtin_amdgcn_wmma_f32_16x16x32_bf16(false, a, false, b1, (short)0, acc1, false, false);
        acc2 = __builtin_amdgcn_wmma_f32_16x16x32_bf16(false, a, false, b2, (short)0, acc2, false, false);
        acc3 = __builtin_amdgcn_wmma_f32_16x16x32_bf16(false, a, false, b3, (short)0, acc3, false, false);
    }

#pragma unroll
    for (int v = 0; v < 8; ++v) {
        const int    mrow = mtile * 16 + v + (hi ? 8 : 0);    // batch index
        const size_t zoff = (size_t)mrow * G_DIM + col;
        const float  zi = acc0[v] + Z_t[zoff];
        const float  zf = acc1[v] + Z_t[zoff + 1 * H_DIM];
        const float  zg = acc2[v] + Z_t[zoff + 2 * H_DIM];
        const float  zo = acc3[v] + Z_t[zoff + 3 * H_DIM];

        const size_t so    = (size_t)mrow * H_DIM + col;
        const float  c_old = rst[mrow] ? 0.0f : c_state[so];
        const float  c_new = sigf(zf) * c_old + sigf(zi) * tanhf_fast(zg);
        const float  h_new = sigf(zo) * tanhf_fast(c_new);

        c_state[so] = c_new;
        h_next[so]  = h_new;
        ys_t[so]    = h_new;
    }
}

// hT, cT -> head of d_out
__global__ void finalize(const float* __restrict__ h, const float* __restrict__ c,
                         float* __restrict__ out) {
    const int i = blockIdx.x * blockDim.x + threadIdx.x;
    out[i]                  = h[i];
    out[B_DIM * H_DIM + i]  = c[i];
}

// ---------------------------------------------------------------------------
extern "C" void kernel_launch(void* const* d_in, const int* in_sizes, int n_in,
                              void* d_out, int out_size, void* d_ws, size_t ws_size,
                              hipStream_t stream) {
    const float*         x    = (const float*)d_in[0];         // [128,64,1024]
    const float*         h0   = (const float*)d_in[1];         // [64,1024]
    const float*         c0   = (const float*)d_in[2];         // [64,1024]
    const float*         Wi   = (const float*)d_in[3];         // [1024,4096]
    const float*         Wh   = (const float*)d_in[4];         // [1024,4096]
    const float*         bias = (const float*)d_in[5];         // [4096]
    const unsigned char* rst  = (const unsigned char*)d_in[6]; // [128,64] bool (1B)

    float* out = (float*)d_out;              // hT(64K) | cT(64K) | ys(128*64*1024)
    float* ys  = out + 2 * B_DIM * H_DIM;

    // workspace layout (all offsets 256B aligned); total ~152 MB
    char*   ws   = (char*)d_ws;
    float*  Z    = (float*)(ws);                                   // 134,217,728 B
    __bf16* Wit  = (__bf16*)(ws + 134217728);                      //   8,388,608 B
    __bf16* Wht  = (__bf16*)(ws + 134217728 + 8388608);            //   8,388,608 B
    float*  hA   = (float*)(ws + 134217728 + 2 * 8388608);         //     262,144 B
    float*  hB   = (float*)(ws + 134217728 + 2 * 8388608 + 262144);//     262,144 B
    float*  cSt  = (float*)(ws + 134217728 + 2 * 8388608 + 2 * 262144);

    // 1) weights -> transposed bf16 (column-major: [4096][1024])
    {
        dim3 g(G_DIM / 16, D_DIM / 16), b(16, 16);
        wt_transpose<<<g, b, 0, stream>>>(Wi, Wit);
        wt_transpose<<<g, b, 0, stream>>>(Wh, Wht);
    }

    // 2) state init
    init_state<<<(B_DIM * H_DIM) / 256, 256, 0, stream>>>(h0, c0, hA, cSt);

    // 3) Z = x @ Wi + b   ([8192,4096] fp32)
    gemm_xwi<<<8192, 128, 0, stream>>>(x, Wit, bias, Z);

    // 4) sequential scan, ping-pong h buffers (read prev / write next)
    float* hbuf[2] = {hA, hB};
    for (int t = 0; t < T_STEPS; ++t) {
        const float*         Z_t   = Z + (size_t)t * B_DIM * G_DIM;
        const unsigned char* rst_t = rst + (size_t)t * B_DIM;
        float*               ys_t  = ys + (size_t)t * B_DIM * H_DIM;
        lstm_step<<<64, 128, 0, stream>>>(Z_t, rst_t, hbuf[t & 1], Wht,
                                          cSt, hbuf[(t + 1) & 1], ys_t);
    }

    // 5) hT (= hbuf[0] after 128 steps), cT
    finalize<<<(B_DIM * H_DIM) / 256, 256, 0, stream>>>(hbuf[0], cSt, out);
}
